// RoutedFFN_56289841381633
// MI455X (gfx1250) — matrix operand order repaired and checked by reading, not compiled, not run on hardware
//
#include <hip/hip_runtime.h>
#include <hip/hip_bf16.h>
#include <math.h>

// Problem dims
#define T_TOK 8192
#define DM    1024
#define DF    4096
#define NH    8
#define NB    4
#define SQ    2048
#define SLICE_T 1024

// GEMM tiling
#define MT 128
#define NT 128
#define KT 64

typedef __bf16 bf16;
typedef __attribute__((ext_vector_type(16))) __bf16 v16bf;
typedef __attribute__((ext_vector_type(8)))  float  v8f;
typedef __attribute__((ext_vector_type(4)))  unsigned int v4u;
typedef __attribute__((ext_vector_type(4)))  int v4i;

// Explicit address-space types so LDS accesses lower to DS ops (not FLAT).
typedef __attribute__((address_space(3))) bf16 sbf16;
typedef __attribute__((address_space(3))) v4u  sv4u;
typedef __attribute__((address_space(1))) v4i* gv4i_p;   // global int4*
typedef __attribute__((address_space(3))) v4i* lv4i_p;   // LDS int4*

union AFrag { v16bf v; v4u u[2]; };
struct Acc { v8f a[2][4]; };

__device__ __forceinline__ float gelu_exact(float x) {
  return 0.5f * x * (1.0f + erff(x * 0.70710678118654752f));
}

__device__ __forceinline__ void zero_acc(Acc& acc) {
#pragma unroll
  for (int i = 0; i < 2; ++i)
#pragma unroll
    for (int j = 0; j < 4; ++j)
      acc.a[i][j] = (v8f){0.f,0.f,0.f,0.f,0.f,0.f,0.f,0.f};
}

// ---- CDNA5 async global->LDS copy (16B per lane), tracked by ASYNCcnt ----
__device__ __forceinline__ void async_copy_b128(const bf16* g, sbf16* l) {
#if __has_builtin(__builtin_amdgcn_global_load_async_to_lds_b128)
  __builtin_amdgcn_global_load_async_to_lds_b128(
      (gv4i_p)(bf16*)g, (lv4i_p)l, 0, 0);
#else
  asm volatile("global_load_async_to_lds_b128 %0, %1, off"
               :: "v"((unsigned)(size_t)l), "v"(g) : "memory");
#endif
}

__device__ __forceinline__ void wait_async0() {
#if __has_builtin(__builtin_amdgcn_s_wait_asynccnt)
  __builtin_amdgcn_s_wait_asynccnt(0);
#else
  asm volatile("s_wait_asynccnt 0x0" ::: "memory");
#endif
}

// ---- staging helpers (256 threads) ----
// A tile: 128 rows x KT=64, async DMA straight into LDS (row-major).
__device__ __forceinline__ void stageA_async(const bf16* __restrict__ Ag,
                                             size_t lda, int k0,
                                             sbf16* dst, int tid) {
#pragma unroll
  for (int i = 0; i < 4; ++i) {
    int c   = tid + i * 256;
    int row = c >> 3;          // KT/8 = 8 chunks per row
    int c8  = c & 7;
    async_copy_b128(Ag + (size_t)row * lda + k0 + c8 * 8,
                    dst + row * KT + c8 * 8);
  }
}

// B tile global -> registers (16B chunks); transpose-scatter later.
__device__ __forceinline__ void loadB_regs(const bf16* __restrict__ Bg,
                                           size_t ldb, int k0,
                                           v4u breg[4], int tid) {
#pragma unroll
  for (int i = 0; i < 4; ++i) {
    int c  = tid + i * 256;
    int kr = c >> 4;           // NT/8 = 16 chunks per k-row
    int n8 = c & 15;
    breg[i] = *(const v4u*)(Bg + (size_t)(k0 + kr) * ldb + n8 * 8);
  }
}

// registers -> LDS transposed Bs[n][k] (ds_store_b16 scatter)
__device__ __forceinline__ void storeB_trans(const v4u breg[4],
                                             sbf16* dst, int tid) {
#pragma unroll
  for (int i = 0; i < 4; ++i) {
    int c  = tid + i * 256;
    int kr = c >> 4;
    int n8 = c & 15;
    union { v4u u; bf16 b[8]; } tmp;
    tmp.u = breg[i];
#pragma unroll
    for (int j = 0; j < 8; ++j) dst[(n8 * 8 + j) * KT + kr] = tmp.b[j];
  }
}

// ---- WMMA compute on one staged KT tile (fragment reads are ds_load_b128) ----
__device__ __forceinline__ void compute_tile(const sbf16* As, const sbf16* Bs,
                                             Acc& acc, int wm, int wn,
                                             int half, int l16) {
#pragma unroll
  for (int kk = 0; kk < KT; kk += 32) {
    AFrag a[2], b[4];
    // A frag: lanes 0-15 M=l16, K={kk..+7, kk+16..+23}; lanes 16-31 K={kk+8..+15, kk+24..+31}
#pragma unroll
    for (int mt = 0; mt < 2; ++mt) {
      const sbf16* p = &As[(wm * 32 + mt * 16 + l16) * KT + kk + half * 8];
      a[mt].u[0] = *(const sv4u*)p;
      a[mt].u[1] = *(const sv4u*)(p + 16);
    }
    // B frag: lanes 0-15 N=l16, K={kk..+7, kk+16..+23}; lanes 16-31 K={kk+8..+15, kk+24..+31}
#pragma unroll
    for (int nt = 0; nt < 4; ++nt) {
      const sbf16* p = &Bs[(wn * 64 + nt * 16 + l16) * KT + kk + half * 8];
      b[nt].u[0] = *(const sv4u*)p;
      b[nt].u[1] = *(const sv4u*)(p + 16);
    }
#pragma unroll
    for (int mt = 0; mt < 2; ++mt)
#pragma unroll
      for (int nt = 0; nt < 4; ++nt)
        acc.a[mt][nt] = __builtin_amdgcn_wmma_f32_16x16x32_bf16(
            false, a[mt].v, false, b[nt].v, (short)0, acc.a[mt][nt],
            false, false);
  }
}

// Double-buffered, async-pipelined 128x128 GEMM.
// A: row-major [rows x Ktot], stride lda; B: row-major [Ktot x cols], stride ldb.
__device__ __forceinline__ void gemm_bf16_128x128(
    const bf16* __restrict__ Ag, size_t lda,
    const bf16* __restrict__ Bg, size_t ldb, int Ktot,
    sbf16* As0, sbf16* As1, sbf16* Bs0, sbf16* Bs1, Acc& acc)
{
  const int tid  = threadIdx.x;
  const int lane = tid & 31;
  const int wid  = tid >> 5;
  const int wm   = wid & 3;
  const int wn   = wid >> 2;
  const int half = lane >> 4;
  const int l16  = lane & 15;

  // Prologue: stage tile 0
  v4u breg[4];
  loadB_regs(Bg, ldb, 0, breg, tid);
  stageA_async(Ag, lda, 0, As0, tid);
  storeB_trans(breg, Bs0, tid);
  wait_async0();
  __syncthreads();

  const int nTiles = Ktot / KT;
  for (int t = 0; t < nTiles; ++t) {
    sbf16* curA = (t & 1) ? As1 : As0;
    sbf16* nxtA = (t & 1) ? As0 : As1;
    sbf16* curB = (t & 1) ? Bs1 : Bs0;
    sbf16* nxtB = (t & 1) ? Bs0 : Bs1;
    const bool more = (t + 1) < nTiles;
    if (more) {
      // issue next tile's traffic before computing current one
      loadB_regs(Bg, ldb, (t + 1) * KT, breg, tid);
      stageA_async(Ag, lda, (t + 1) * KT, nxtA, tid);
    }
    compute_tile(curA, curB, acc, wm, wn, half, l16);
    if (more) {
      storeB_trans(breg, nxtB, tid);  // loadcnt wait lands here, post-compute
      wait_async0();
    }
    __syncthreads();
  }
}

// ---------------- conversion ----------------
__global__ void cvt_f32_bf16(const float* __restrict__ src,
                             bf16* __restrict__ dst, size_t n) {
  size_t stride = (size_t)gridDim.x * blockDim.x * 4;
  for (size_t i = ((size_t)blockIdx.x * blockDim.x + threadIdx.x) * 4;
       i < n; i += stride) {
    float4 f = *(const float4*)(src + i);
    union { bf16 b[4]; unsigned long long u; } t;
    t.b[0] = (bf16)f.x; t.b[1] = (bf16)f.y;
    t.b[2] = (bf16)f.z; t.b[3] = (bf16)f.w;
    *(unsigned long long*)(dst + i) = t.u;
  }
}

// ---------------- sproj = routing_state @ Ws + bs ----------------
__global__ __launch_bounds__(256) void sproj_kernel(
    const float* __restrict__ rs, const float* __restrict__ Ws,
    const float* __restrict__ bs, float* __restrict__ sp) {
  int d = blockIdx.x * 256 + threadIdx.x;
  int b = blockIdx.y;
  float s = bs[d];
  for (int k = 0; k < DM; ++k) s += rs[b * DM + k] * Ws[(size_t)k * DM + d];
  sp[b * DM + d] = s;
}

// ---------------- cg = gelu(x@Wi + bi + sproj[b]) (bf16) ----------------
__global__ __launch_bounds__(256) void gproj_kernel(
    const bf16* __restrict__ xb, const bf16* __restrict__ wib,
    const float* __restrict__ sp, const float* __restrict__ bi,
    bf16* __restrict__ cg)
{
  __shared__ bf16 AsArr[2][MT * KT];
  __shared__ bf16 BsArr[2][NT * KT];
  sbf16* As0 = (sbf16*)AsArr[0]; sbf16* As1 = (sbf16*)AsArr[1];
  sbf16* Bs0 = (sbf16*)BsArr[0]; sbf16* Bs1 = (sbf16*)BsArr[1];
  const int m0 = blockIdx.y * MT;
  const int n0 = blockIdx.x * NT;
  Acc acc; zero_acc(acc);
  gemm_bf16_128x128(xb + (size_t)m0 * DM, DM, wib + n0, DM, DM,
                    As0, As1, Bs0, Bs1, acc);

  const int tid = threadIdx.x, lane = tid & 31, wid = tid >> 5;
  const int wm = wid & 3, wn = wid >> 2, half = lane >> 4, l16 = lane & 15;
#pragma unroll
  for (int mt = 0; mt < 2; ++mt)
#pragma unroll
    for (int nt = 0; nt < 4; ++nt) {
      int n = n0 + wn * 64 + nt * 16 + l16;
      float bb = bi[n];
#pragma unroll
      for (int v = 0; v < 8; ++v) {
        int m = m0 + wm * 32 + mt * 16 + half * 8 + v;
        int b = m / SQ;
        float val = acc.a[mt][nt][v] + bb + sp[b * DM + n];
        cg[(size_t)m * DM + n] = (bf16)gelu_exact(val);
      }
    }
}

// ---------------- logits = cg @ Wg + bg ; softmax -> weights ----------------
__global__ __launch_bounds__(256) void gate_kernel(
    const bf16* __restrict__ cg, const float* __restrict__ Wg,
    const float* __restrict__ bg, float* __restrict__ wts)
{
  __shared__ float red[NH * 256];
  int t = blockIdx.x;
  float p[NH];
#pragma unroll
  for (int hh = 0; hh < NH; ++hh) p[hh] = 0.f;
  for (int d = threadIdx.x; d < DM; d += 256) {
    float g = (float)cg[(size_t)t * DM + d];
#pragma unroll
    for (int hh = 0; hh < NH; ++hh) p[hh] += g * Wg[d * NH + hh];
  }
#pragma unroll
  for (int hh = 0; hh < NH; ++hh) red[hh * 256 + threadIdx.x] = p[hh];
  __syncthreads();
  for (int s2 = 128; s2 > 0; s2 >>= 1) {
    if (threadIdx.x < (unsigned)s2) {
#pragma unroll
      for (int hh = 0; hh < NH; ++hh)
        red[hh * 256 + threadIdx.x] += red[hh * 256 + threadIdx.x + s2];
    }
    __syncthreads();
  }
  if (threadIdx.x == 0) {
    float l[NH], mx = -1e30f;
#pragma unroll
    for (int hh = 0; hh < NH; ++hh) {
      l[hh] = red[hh * 256] + bg[hh];
      mx = fmaxf(mx, l[hh]);
    }
    float sum = 0.f;
#pragma unroll
    for (int hh = 0; hh < NH; ++hh) { l[hh] = expf(l[hh] - mx); sum += l[hh]; }
    float inv = 1.f / sum;
#pragma unroll
    for (int hh = 0; hh < NH; ++hh) wts[(size_t)t * NH + hh] = l[hh] * inv;
  }
}

// ---------------- hidden = gelu(x@W1[h] + b1[h]) (bf16, slice-local) ----------------
__global__ __launch_bounds__(256) void ffn1_kernel(
    const bf16* __restrict__ xb, const bf16* __restrict__ w1b,
    const float* __restrict__ b1, bf16* __restrict__ hid, int t_base)
{
  __shared__ bf16 AsArr[2][MT * KT];
  __shared__ bf16 BsArr[2][NT * KT];
  sbf16* As0 = (sbf16*)AsArr[0]; sbf16* As1 = (sbf16*)AsArr[1];
  sbf16* Bs0 = (sbf16*)BsArr[0]; sbf16* Bs1 = (sbf16*)BsArr[1];
  const int h  = blockIdx.z;
  const int m0 = blockIdx.y * MT;   // slice-local token
  const int n0 = blockIdx.x * NT;
  Acc acc; zero_acc(acc);
  gemm_bf16_128x128(xb + (size_t)(t_base + m0) * DM, DM,
                    w1b + (size_t)h * DM * DF + n0, DF, DM,
                    As0, As1, Bs0, Bs1, acc);

  const int tid = threadIdx.x, lane = tid & 31, wid = tid >> 5;
  const int wm = wid & 3, wn = wid >> 2, half = lane >> 4, l16 = lane & 15;
#pragma unroll
  for (int mt = 0; mt < 2; ++mt)
#pragma unroll
    for (int nt = 0; nt < 4; ++nt) {
      int n = n0 + wn * 64 + nt * 16 + l16;
      float bb = b1[h * DF + n];
#pragma unroll
      for (int v = 0; v < 8; ++v) {
        int m = m0 + wm * 32 + mt * 16 + half * 8 + v;
        float g = gelu_exact(acc.a[mt][nt][v] + bb);
        hid[((size_t)m * NH + h) * DF + n] = (bf16)g;
      }
    }
}

// ---------------- out = sum_h w[t,h]*(hidden@W2[h] + b2[h]) ----------------
__global__ __launch_bounds__(256) void ffn2_kernel(
    const bf16* __restrict__ hid, const bf16* __restrict__ w2b,
    const float* __restrict__ b2, const float* __restrict__ wts,
    float* __restrict__ out, int t_base)
{
  __shared__ bf16 AsArr[2][MT * KT];
  __shared__ bf16 BsArr[2][NT * KT];
  sbf16* As0 = (sbf16*)AsArr[0]; sbf16* As1 = (sbf16*)AsArr[1];
  sbf16* Bs0 = (sbf16*)BsArr[0]; sbf16* Bs1 = (sbf16*)BsArr[1];
  const int m0 = blockIdx.y * MT;   // slice-local token
  const int n0 = blockIdx.x * NT;
  const int tid = threadIdx.x, lane = tid & 31, wid = tid >> 5;
  const int wm = wid & 3, wn = wid >> 2, half = lane >> 4, l16 = lane & 15;

  v8f outa[2][4];
#pragma unroll
  for (int i = 0; i < 2; ++i)
#pragma unroll
    for (int j = 0; j < 4; ++j)
      outa[i][j] = (v8f){0.f,0.f,0.f,0.f,0.f,0.f,0.f,0.f};

  for (int h = 0; h < NH; ++h) {
    Acc acc; zero_acc(acc);
    gemm_bf16_128x128(hid + ((size_t)m0 * NH + h) * DF, (size_t)NH * DF,
                      w2b + (size_t)h * DF * DM + n0, DM, DF,
                      As0, As1, Bs0, Bs1, acc);
#pragma unroll
    for (int mt = 0; mt < 2; ++mt) {
      float wv[8];
#pragma unroll
      for (int v = 0; v < 8; ++v) {
        int m = t_base + m0 + wm * 32 + mt * 16 + half * 8 + v;
        wv[v] = wts[(size_t)m * NH + h];
      }
#pragma unroll
      for (int nt = 0; nt < 4; ++nt) {
        int n = n0 + wn * 64 + nt * 16 + l16;
        float bb = b2[h * DM + n];
#pragma unroll
        for (int v = 0; v < 8; ++v)
          outa[mt][nt][v] += wv[v] * (acc.a[mt][nt][v] + bb);
      }
    }
  }
#pragma unroll
  for (int mt = 0; mt < 2; ++mt)
#pragma unroll
    for (int nt = 0; nt < 4; ++nt) {
      int n = n0 + wn * 64 + nt * 16 + l16;
#pragma unroll
      for (int v = 0; v < 8; ++v) {
        int m = t_base + m0 + wm * 32 + mt * 16 + half * 8 + v;
        out[(size_t)m * DM + n] = outa[mt][nt][v];
      }
    }
}

extern "C" void kernel_launch(void* const* d_in, const int* in_sizes, int n_in,
                              void* d_out, int out_size, void* d_ws, size_t ws_size,
                              hipStream_t stream)
{
  (void)in_sizes; (void)n_in; (void)out_size; (void)ws_size;
  const float* x  = (const float*)d_in[0];
  const float* rs = (const float*)d_in[1];
  const float* W1 = (const float*)d_in[2];
  const float* b1 = (const float*)d_in[3];
  const float* W2 = (const float*)d_in[4];
  const float* b2 = (const float*)d_in[5];
  const float* Ws = (const float*)d_in[6];
  const float* bs = (const float*)d_in[7];
  const float* Wi = (const float*)d_in[8];
  const float* bi = (const float*)d_in[9];
  const float* Wg = (const float*)d_in[10];
  const float* bg = (const float*)d_in[11];

  float* out = (float*)d_out;
  float* wts = out + (size_t)T_TOK * DM;   // weights output region

  char* ws = (char*)d_ws;
  size_t off = 0;
  bf16* xb  = (bf16*)(ws + off); off += (size_t)T_TOK * DM * 2;      // 16 MB
  bf16* w1b = (bf16*)(ws + off); off += (size_t)NH * DM * DF * 2;    // 64 MB
  bf16* w2b = (bf16*)(ws + off); off += (size_t)NH * DF * DM * 2;    // 64 MB
  bf16* wib = (bf16*)(ws + off); off += (size_t)DM * DM * 2;         // 2 MB
  float* sp = (float*)(ws + off); off += (size_t)NB * DM * 4;        // 16 KB
  bf16* cg  = (bf16*)(ws + off); off += (size_t)T_TOK * DM * 2;      // 16 MB
  bf16* hid = (bf16*)(ws + off); off += (size_t)SLICE_T * NH * DF * 2; // 64 MB

  cvt_f32_bf16<<<2048, 256, 0, stream>>>(x,  xb,  (size_t)T_TOK * DM);
  cvt_f32_bf16<<<4096, 256, 0, stream>>>(W1, w1b, (size_t)NH * DM * DF);
  cvt_f32_bf16<<<4096, 256, 0, stream>>>(W2, w2b, (size_t)NH * DF * DM);
  cvt_f32_bf16<<<1024, 256, 0, stream>>>(Wi, wib, (size_t)DM * DM);

  sproj_kernel<<<dim3(DM / 256, NB), 256, 0, stream>>>(rs, Ws, bs, sp);
  gproj_kernel<<<dim3(DM / NT, T_TOK / MT), 256, 0, stream>>>(xb, wib, sp, bi, cg);
  gate_kernel<<<T_TOK, 256, 0, stream>>>(cg, Wg, bg, wts);

  for (int s = 0; s < T_TOK / SLICE_T; ++s) {
    int t_base = s * SLICE_T;
    ffn1_kernel<<<dim3(DF / NT, SLICE_T / MT, NH), 256, 0, stream>>>(
        xb, w1b, b1, hid, t_base);
    ffn2_kernel<<<dim3(DM / NT, SLICE_T / MT), 256, 0, stream>>>(
        hid, w2b, b2, wts, out, t_base);
  }
}
